// RoughSetFeatureSelector_23794118820466
// MI455X (gfx1250) — matrix operand order repaired and checked by reading, not compiled, not run on hardware
//
#include <hip/hip_runtime.h>

// Problem constants (match reference)
#define NN 8192
#define DD 256
#define NSPLIT 8            // column splits of the Gram matrix
#define JCHUNK (NN / NSPLIT)

typedef __attribute__((ext_vector_type(16))) _Float16 v16h;
typedef __attribute__((ext_vector_type(8)))  float    v8f;

union F16Frag { v16h v; uint4 q[2]; };

// ---------------------------------------------------------------- weights ---
__global__ void rs_weights_kernel(const float* __restrict__ theta,
                                  float* __restrict__ weights) {
    int d = threadIdx.x;
    weights[d] = 1.0f / (1.0f + __expf(-theta[d]));
}

// ----------------------------------------------------- xw + f16 copy + |.|² --
__global__ void rs_xw_kernel(const float* __restrict__ x,
                             const float* __restrict__ weights,
                             float* __restrict__ xw_out,
                             _Float16* __restrict__ xw16,
                             float* __restrict__ sqn) {
    __shared__ float red[DD];
    int i = blockIdx.x;
    int d = threadIdx.x;
    float v = x[(size_t)i * DD + d] * weights[d];
    xw_out[(size_t)i * DD + d] = v;
    xw16[(size_t)i * DD + d] = (_Float16)v;
    red[d] = v * v;
    __syncthreads();
    for (int s = DD / 2; s > 0; s >>= 1) {
        if (d < s) red[d] += red[d + s];
        __syncthreads();
    }
    if (d == 0) sqn[i] = red[0];
}

// ------------------------------------------------------------------- zero ---
__global__ void rs_zero_kernel(float* __restrict__ num, float* __restrict__ den) {
    int i = blockIdx.x * blockDim.x + threadIdx.x;
    num[i] = 0.0f;
    den[i] = 0.0f;
}

// ----------------------------------------- fused WMMA Gram + rough-set loss --
// Block: 256 threads = 8 waves. Wave w owns rows [blockIdx.x*128 + w*16, +16).
// Column split blockIdx.y covers JCHUNK columns, 64 per iteration (4 n-tiles).
// B panels (64 rows x 256 halves = 32 KB) are double-buffered in LDS and
// filled with GLOBAL_LOAD_ASYNC_TO_LDS_B128 (ASYNCcnt / s_wait_asynccnt).
// B fragments are software-pipelined across k-steps (register double buffer)
// so ds_load latency hides under the WMMA stream.
__global__ __launch_bounds__(256)
void rs_gram_loss_kernel(const _Float16* __restrict__ xw16,
                         const float* __restrict__ sqn,
                         const int* __restrict__ labels,
                         float* __restrict__ num,
                         float* __restrict__ den) {
    __shared__ _Float16 panel[2][64 * DD];   // 2 x 32 KB

    const int wid  = threadIdx.x >> 5;
    const int lane = threadIdx.x & 31;
    const int hi   = lane >> 4;   // half-wave (0/1)
    const int l15  = lane & 15;

    const int iBase = blockIdx.x * 128 + wid * 16;
    const int jBase = blockIdx.y * JCHUNK;

    // ---- Preload all 8 A k-step fragments (rows fixed for entire kernel). --
    // CDNA5 16-bit A 16x32 layout: lane m=l%16; elems 0..7 -> K = 8*hi + e,
    // elems 8..15 -> K = 16 + 8*hi + (e-8).
    const _Float16* arow = xw16 + (size_t)(iBase + l15) * DD;
    F16Frag afrag[8];
#pragma unroll
    for (int kk = 0; kk < 8; ++kk) {
        const int k0 = kk * 32;
        afrag[kk].q[0] = *(const uint4*)(arow + k0 + 8 * hi);
        afrag[kk].q[1] = *(const uint4*)(arow + k0 + 16 + 8 * hi);
    }

    // ---- Per-lane info for the 8 C-rows this lane owns (C/D layout:
    // lane holds col n=l%16, rows M = v + 8*hi). Same rows all kernel long. --
    const int rowB = iBase + 8 * hi;
    float sqn_i[8];
    int   lab_i[8];
#pragma unroll
    for (int v = 0; v < 8; ++v) {
        sqn_i[v] = sqn[rowB + v];
        lab_i[v] = labels[rowB + v];
    }

    float numP[8], denP[8];
#pragma unroll
    for (int v = 0; v < 8; ++v) { numP[v] = 0.0f; denP[v] = 0.0f; }

    // ---- Async-copy of one 32 KB B panel: 8 x b128 per thread. -------------
    // Panel rows jb..jb+63 are contiguous in xw16, so the copy is flat.
    auto issue_panel = [&](int buf, int jb) {
        const char* gsrc = (const char*)(xw16 + (size_t)jb * DD);
        unsigned    ldsb = (unsigned)(size_t)(&panel[buf][0]);
#pragma unroll
        for (int i = 0; i < 8; ++i) {
            unsigned    off = (unsigned)threadIdx.x * 16u + (unsigned)i * 4096u;
            unsigned    ldsd = ldsb + off;
            const char* g    = gsrc + off;
            asm volatile("global_load_async_to_lds_b128 %0, %1, off"
                         :: "v"(ldsd), "v"(g)
                         : "memory");
        }
    };

    // B-fragment loader: CDNA5 16-bit B 32x16 layout: lane holds col n=l%16,
    // elems 0..7 -> K = 8*hi + e, elems 8..15 -> K = 16 + 8*hi + (e-8).
    auto load_bfrag = [&](F16Frag& f, const _Float16* base, int t, int k0) {
        const _Float16* brow =
            base + (size_t)(t * 16 + l15) * DD + k0 + 8 * hi;
        f.q[0] = *(const uint4*)(brow);
        f.q[1] = *(const uint4*)(brow + 16);
    };

    // Prologue: fill buffer 0 with the first panel.
    issue_panel(0, jBase);

    // ---- Column loop: 4 n-tiles (64 cols) per iteration, double-buffered. --
    for (int jt = 0; jt < JCHUNK; jt += 64) {
        const int jb  = jBase + jt;
        const int cur = (jt >> 6) & 1;

        // My async loads for panel[cur] are complete.
        asm volatile("s_wait_asynccnt 0x0" ::: "memory");
        // Everyone's panel[cur] loads landed; everyone finished reading
        // panel[1-cur] (their previous compute precedes this barrier).
        __syncthreads();

        // Overlap: start fetching the next panel into the idle buffer.
        if (jt + 64 < JCHUNK) {
            issue_panel(1 - cur, jb + 64);
        }

        const _Float16* pcur = &panel[cur][0];

        v8f acc[4];
#pragma unroll
        for (int t = 0; t < 4; ++t) acc[t] = (v8f){0,0,0,0,0,0,0,0};

        // Software-pipelined k-loop: load k-step kk+1's B fragments into the
        // other register buffer while issuing k-step kk's 4 WMMAs, so the
        // ds_load latency hides under the matrix ops.
        F16Frag bbuf[2][4];
#pragma unroll
        for (int t = 0; t < 4; ++t) load_bfrag(bbuf[0][t], pcur, t, 0);

#pragma unroll
        for (int kk = 0; kk < 8; ++kk) {
            const int pb = kk & 1;
            if (kk < 7) {
#pragma unroll
                for (int t = 0; t < 4; ++t)
                    load_bfrag(bbuf[pb ^ 1][t], pcur, t, (kk + 1) * 32);
            }
#pragma unroll
            for (int t = 0; t < 4; ++t) {
                acc[t] = __builtin_amdgcn_wmma_f32_16x16x32_f16(
                    false, afrag[kk].v, false, bbuf[pb][t].v,
                    (short)0, acc[t], false, false);
            }
        }

        // ---- Fused epilogue: dist -> R -> enemy-weighted row sums. ---------
        // Branchless so EXEC stays all-1s for the next iteration's WMMAs.
#pragma unroll
        for (int t = 0; t < 4; ++t) {
            const int j = jb + t * 16 + l15;
            const float sqn_j = sqn[j];
            const int   lab_j = labels[j];
#pragma unroll
            for (int v = 0; v < 8; ++v) {
                float g = acc[t][v];
                float dist = fmaxf(fmaf(-2.0f, g, sqn_i[v] + sqn_j), 0.0f);
                float R = __expf(-0.5f * dist);
                float m = (lab_i[v] != lab_j) ? 1.0f : 0.0f;
                float w = __expf(10.0f * R) * m;
                numP[v] += R * w;
                denP[v] += w;
            }
        }
    }

    // ---- Cross-lane reduction: 16 lanes of a half-wave share each row. -----
#pragma unroll
    for (int v = 0; v < 8; ++v) {
        float n = numP[v];
        float d = denP[v];
        for (int m = 1; m < 16; m <<= 1) {
            n += __shfl_xor(n, m, 32);
            d += __shfl_xor(d, m, 32);
        }
        if (l15 == 0) {
            atomicAdd(&num[rowB + v], n);
            atomicAdd(&den[rowB + v], d);
        }
    }
}

// --------------------------------------------------------------- finalize ---
__global__ void rs_finalize_kernel(const float* __restrict__ num,
                                   const float* __restrict__ den,
                                   float* __restrict__ loss_out) {
    __shared__ float red[256];
    float acc = 0.0f;
    for (int i = threadIdx.x; i < NN; i += 256) {
        float d = den[i];
        // den > 0 <=> row has an enemy (each enemy contributes exp(10R) >= 1)
        float mu = (d > 0.0f) ? (1.0f - num[i] / (d + 1e-8f)) : 1.0f;
        acc += mu;
    }
    red[threadIdx.x] = acc;
    __syncthreads();
    for (int s = 128; s > 0; s >>= 1) {
        if (threadIdx.x < s) red[threadIdx.x] += red[threadIdx.x + s];
        __syncthreads();
    }
    if (threadIdx.x == 0) loss_out[0] = 1.0f - red[0] / (float)NN;
}

// ------------------------------------------------------------------ launch ---
extern "C" void kernel_launch(void* const* d_in, const int* in_sizes, int n_in,
                              void* d_out, int out_size, void* d_ws, size_t ws_size,
                              hipStream_t stream) {
    const float* x      = (const float*)d_in[0];
    const int*   labels = (const int*)d_in[1];
    const float* theta  = (const float*)d_in[2];

    float* out      = (float*)d_out;
    float* xw_out   = out;                        // [N, D] f32
    float* w_out    = out + (size_t)NN * DD;      // [D]    f32
    float* loss_out = w_out + DD;                 // [1]    f32

    char* ws = (char*)d_ws;
    _Float16* xw16 = (_Float16*)ws;                           // 4 MB
    float*    sqn  = (float*)(ws + (size_t)NN * DD * 2);      // 32 KB
    float*    num  = sqn + NN;                                // 32 KB
    float*    den  = num + NN;                                // 32 KB

    rs_weights_kernel<<<1, DD, 0, stream>>>(theta, w_out);
    rs_xw_kernel<<<NN, DD, 0, stream>>>(x, w_out, xw_out, xw16, sqn);
    rs_zero_kernel<<<NN / 256, 256, 0, stream>>>(num, den);

    dim3 grid(NN / 128, NSPLIT);
    rs_gram_loss_kernel<<<grid, 256, 0, stream>>>(xw16, sqn, labels, num, den);

    rs_finalize_kernel<<<1, 256, 0, stream>>>(num, den, loss_out);
}
